// Graphormer_36292473651599
// MI455X (gfx1250) — matrix operand order repaired and checked by reading, not compiled, not run on hardware
//
#include <hip/hip_runtime.h>
#include <hip/hip_bf16.h>

// ---------------------------------------------------------------------------
// Graphormer forward for MI455X (gfx1250).  All heavy GEMMs run through
// v_wmma_f32_16x16x32_bf16 (bf16 inputs, f32 accumulate).  wave32 layouts per
// CDNA5 ISA 7.12.2.
// ---------------------------------------------------------------------------

#define G_   128
#define N1_  128
#define M_   16384      // G*N1 rows of the residual stream
#define F_   128
#define H_   512
#define FFN_ 2048
#define NH_  16
#define DH_  32
#define L_   4
#define E_   262144
#define P_   2097152    // G*N1*N1
#define MAXD_ 5

typedef __attribute__((ext_vector_type(16))) __bf16 bf16x16;
typedef __attribute__((ext_vector_type(8)))  float  floatx8;

__device__ __forceinline__ __bf16 f2bf(float f) {
  union { float f; unsigned u; } v; v.f = f;
  unsigned r = v.u + 0x7FFFu + ((v.u >> 16) & 1u);   // round-to-nearest-even
  unsigned short h = (unsigned short)(r >> 16);
  __bf16 o; __builtin_memcpy(&o, &h, 2); return o;
}

__device__ __forceinline__ floatx8 wmma_bf16(bf16x16 a, bf16x16 b, floatx8 c) {
  return __builtin_amdgcn_wmma_f32_16x16x32_bf16(
      /*neg_a=*/false, a, /*neg_b=*/false, b,
      /*c_mod=*/(short)0, c, /*reuse_a=*/false, /*reuse_b=*/false);
}

__device__ __forceinline__ float gelu_f(float x) {
  float x3 = x * x * x;
  return 0.5f * x * (1.f + tanhf(0.7978845608028654f * (x + 0.044715f * x3)));
}

// ---------------------------------------------------------------------------
// Templated WMMA GEMM:  C[M,N] = A[M,K](bf16) @ Bt[N,K](bf16, pre-transposed)
// Block: 256 threads = 8 waves, 64x64 tile; wave (wm 0..3, wn 0..1) owns
// rows wm*16..+16, cols wn*32..+32 (two 16x16 accumulators).
// MODE 0: f32 store + bias            (embedding)
// MODE 1: bf16 store + bias           (Q, K)
// MODE 2: bf16 transposed store +bias (V -> Vt[N][M])
// MODE 3: f32 residual += acc + bias  (O-proj, FFN2)
// MODE 4: bf16 store + bias + GELU    (FFN1)
// ---------------------------------------------------------------------------
template<int MODE>
__global__ __launch_bounds__(256)
void gemm_wmma(const __bf16* __restrict__ A, const __bf16* __restrict__ Bt,
               const float* __restrict__ bias,
               float* __restrict__ Cf, __bf16* __restrict__ Cb,
               int M, int N, int K)
{
  const int lane = threadIdx.x & 31;
  const int wid  = threadIdx.x >> 5;
  const int wm = wid & 3, wn = wid >> 2;
  const int rowBase = blockIdx.x * 64 + wm * 16;
  const int colBase = blockIdx.y * 64 + wn * 32;

  // A fragment addressing (16-bit A 16x32): lane = m (0..15) in each half,
  // k-range: lanes 0-15 -> {0..15}; lanes 16-31 -> {16..31}
  const int am  = lane & 15;
  const int akb = (lane >> 4) * 16;
  // B fragment (32x16): lane = n, k = (lane>>4)*16 + e
  const int bn  = lane & 15;
  const int bkb = (lane >> 4) * 16;

  const __bf16* Ap = A  + (size_t)(rowBase + am) * K + akb;
  const __bf16* B0 = Bt + (size_t)(colBase + bn) * K + bkb;
  const __bf16* B1 = Bt + (size_t)(colBase + 16 + bn) * K + bkb;

  floatx8 d0 = {}; floatx8 d1 = {};
  for (int k0 = 0; k0 < K; k0 += 32) {
    bf16x16 a, b0, b1;
    ((uint4*)&a )[0] = *(const uint4*)(Ap + k0);
    ((uint4*)&a )[1] = *(const uint4*)(Ap + k0 + 8);
    ((uint4*)&b0)[0] = *(const uint4*)(B0 + k0);
    ((uint4*)&b0)[1] = *(const uint4*)(B0 + k0 + 8);
    ((uint4*)&b1)[0] = *(const uint4*)(B1 + k0);
    ((uint4*)&b1)[1] = *(const uint4*)(B1 + k0 + 8);
    d0 = wmma_bf16(a, b0, d0);
    d1 = wmma_bf16(a, b1, d1);
  }

  // D layout: lane holds column n = lane&15 (+tile base); rows m = (lane>>4)*8+e
  const int cn0 = colBase + (lane & 15);
  const int cn1 = cn0 + 16;
  const int mb  = rowBase + ((lane >> 4) * 8);
  const float bia0 = bias ? bias[cn0] : 0.f;
  const float bia1 = bias ? bias[cn1] : 0.f;
#pragma unroll
  for (int e = 0; e < 8; ++e) {
    const int m = mb + e;
    float v0 = d0[e] + bia0;
    float v1 = d1[e] + bia1;
    if (MODE == 0) {
      Cf[(size_t)m * N + cn0] = v0;               Cf[(size_t)m * N + cn1] = v1;
    } else if (MODE == 1) {
      Cb[(size_t)m * N + cn0] = f2bf(v0);         Cb[(size_t)m * N + cn1] = f2bf(v1);
    } else if (MODE == 2) {
      Cb[(size_t)cn0 * M + m] = f2bf(v0);         Cb[(size_t)cn1 * M + m] = f2bf(v1);
    } else if (MODE == 3) {
      Cf[(size_t)m * N + cn0] += v0;              Cf[(size_t)m * N + cn1] += v1;
    } else { // MODE 4
      Cb[(size_t)m * N + cn0] = f2bf(gelu_f(v0)); Cb[(size_t)m * N + cn1] = f2bf(gelu_f(v1));
    }
  }
}

// ---------------------------------------------------------------------------
// Attention: one block per (graph, head), 4 waves.
//   S(128x128) = Q_h(128x32) @ K_h^T  via WMMA (K = 32 = DH, one step)
//   f32 softmax (+ structural bias) in LDS, rewritten as bf16 P
//   O(128x32)  = P @ V_h via WMMA (V pre-transposed: Vt[H][16384])
// ---------------------------------------------------------------------------
__global__ __launch_bounds__(128)
void attn_kernel(const __bf16* __restrict__ Q, const __bf16* __restrict__ Kb,
                 const __bf16* __restrict__ Vt, const float* __restrict__ bias,
                 __bf16* __restrict__ O)
{
  __shared__ float S[N1_ * N1_];               // 64 KB
  __bf16* P = (__bf16*)S;                      // overlay; P row stride = 256 bf16
                                               // (P row m lives inside S row m -> no cross-thread clobber)
  const int g = blockIdx.x;
  const int h = blockIdx.y;
  const int lane = threadIdx.x & 31;
  const int wid  = threadIdx.x >> 5;           // 0..3, wave owns rows wid*32..+32
  const int nodeBase = g * N1_;
  const int am = lane & 15, akb = (lane >> 4) * 16;
  const int bn = lane & 15, bkb = (lane >> 4) * 16;

  // ---- phase 1: raw scores into LDS ----
  for (int rt = 0; rt < 2; ++rt) {
    const int row0 = wid * 32 + rt * 16;
    const __bf16* Ap = Q + (size_t)(nodeBase + row0 + am) * H_ + h * DH_ + akb;
    bf16x16 a;
    ((uint4*)&a)[0] = *(const uint4*)(Ap);
    ((uint4*)&a)[1] = *(const uint4*)(Ap + 8);
    for (int ct = 0; ct < 8; ++ct) {
      const int col0 = ct * 16;
      const __bf16* Bp = Kb + (size_t)(nodeBase + col0 + bn) * H_ + h * DH_ + bkb;
      bf16x16 b;
      ((uint4*)&b)[0] = *(const uint4*)(Bp);
      ((uint4*)&b)[1] = *(const uint4*)(Bp + 8);
      floatx8 d = {};
      d = wmma_bf16(a, b, d);
      const int n = col0 + (lane & 15);
      const int mr = row0 + ((lane >> 4) * 8);
#pragma unroll
      for (int e = 0; e < 8; ++e) S[(mr + e) * N1_ + n] = d[e];
    }
  }
  __syncthreads();

  // ---- phase 2: softmax row-per-thread ----
  {
    const int m = threadIdx.x;                       // 0..127
    const float inv_scale = 0.17677669529663687f;    // 1/sqrt(32)
    const float* brow = bias + (size_t)g * (N1_ * N1_) + m * N1_;
    float mx = -3.4e38f;
    for (int n = 0; n < N1_; ++n) {
      float v = S[m * N1_ + n] * inv_scale + brow[n];
      S[m * N1_ + n] = v;
      mx = fmaxf(mx, v);
    }
    float sum = 0.f;
    for (int n = 0; n < N1_; ++n) {
      float e = __expf(S[m * N1_ + n] - mx);
      S[m * N1_ + n] = e;
      sum += e;
    }
    const float inv = 1.f / sum;
    for (int n = 0; n < N1_; ++n)                    // in-place f32 -> bf16 (own row only)
      P[m * 256 + n] = f2bf(S[m * N1_ + n] * inv);
  }
  __syncthreads();

  // ---- phase 3: O = P @ V_h ----
  for (int rt = 0; rt < 2; ++rt) {
    const int row0 = wid * 32 + rt * 16;
    floatx8 d0 = {}, d1 = {};
    for (int k0 = 0; k0 < N1_; k0 += 32) {
      const __bf16* Ap = P + (row0 + am) * 256 + k0 + akb;   // LDS (ds_load)
      bf16x16 a;
      ((uint4*)&a)[0] = *(const uint4*)(Ap);
      ((uint4*)&a)[1] = *(const uint4*)(Ap + 8);
      const __bf16* B0 = Vt + (size_t)(h * DH_ + bn)      * M_ + nodeBase + k0 + bkb;
      const __bf16* B1 = Vt + (size_t)(h * DH_ + 16 + bn) * M_ + nodeBase + k0 + bkb;
      bf16x16 b0, b1;
      ((uint4*)&b0)[0] = *(const uint4*)(B0);
      ((uint4*)&b0)[1] = *(const uint4*)(B0 + 8);
      ((uint4*)&b1)[0] = *(const uint4*)(B1);
      ((uint4*)&b1)[1] = *(const uint4*)(B1 + 8);
      d0 = wmma_bf16(a, b0, d0);
      d1 = wmma_bf16(a, b1, d1);
    }
    const int n0 = h * DH_ + (lane & 15);
    const int n1 = n0 + 16;
    const int mr = row0 + ((lane >> 4) * 8);
#pragma unroll
    for (int e = 0; e < 8; ++e) {
      O[(size_t)(nodeBase + mr + e) * H_ + n0] = f2bf(d0[e]);
      O[(size_t)(nodeBase + mr + e) * H_ + n1] = f2bf(d1[e]);
    }
  }
}

// ---------------------------------------------------------------------------
// LayerNorm over H=512, one wave per row (16 elems/lane, shfl_xor reduction)
// ---------------------------------------------------------------------------
__global__ __launch_bounds__(256)
void layernorm_bf16(const float* __restrict__ X, const float* __restrict__ s,
                    const float* __restrict__ b, __bf16* __restrict__ Y)
{
  const int row  = blockIdx.x * 8 + (threadIdx.x >> 5);
  const int lane = threadIdx.x & 31;
  const float* xr = X + (size_t)row * H_;
  float v[16];
  float sum = 0.f;
#pragma unroll
  for (int i = 0; i < 16; ++i) { v[i] = xr[lane + i * 32]; sum += v[i]; }
#pragma unroll
  for (int o = 16; o >= 1; o >>= 1) sum += __shfl_xor(sum, o, 32);
  const float mu = sum * (1.f / (float)H_);
  float var = 0.f;
#pragma unroll
  for (int i = 0; i < 16; ++i) { float d = v[i] - mu; var += d * d; }
#pragma unroll
  for (int o = 16; o >= 1; o >>= 1) var += __shfl_xor(var, o, 32);
  const float rstd = rsqrtf(var * (1.f / (float)H_) + 1e-5f);
  __bf16* yr = Y + (size_t)row * H_;
#pragma unroll
  for (int i = 0; i < 16; ++i) {
    const int c = lane + i * 32;
    yr[c] = f2bf((v[i] - mu) * rstd * s[c] + b[c]);
  }
}

// ---------------------------------------------------------------------------
// Small helper kernels
// ---------------------------------------------------------------------------
__global__ void zero_i32(int* p, int n) {
  int i = blockIdx.x * 256 + threadIdx.x; if (i < n) p[i] = 0;
}

__global__ void degrees_k(const int* __restrict__ ei, int* __restrict__ ind,
                          int* __restrict__ outd) {
  int e = blockIdx.x * 256 + threadIdx.x; if (e >= E_) return;
  atomicAdd(&outd[ei[e]], 1);          // edge_index[0] = src -> out degree
  atomicAdd(&ind [ei[E_ + e]], 1);     // edge_index[1] = dst -> in degree
}

__global__ void add_deg_emb(float* __restrict__ Hh, const int* __restrict__ ind,
                            const int* __restrict__ outd,
                            const float* __restrict__ z_in,
                            const float* __restrict__ z_out) {
  int idx = blockIdx.x * 256 + threadIdx.x;          // M_*H_ threads
  int node = idx >> 9, c = idx & 511;
  int di = ind[node];  di = di < 63 ? di : 63;
  int dd = outd[node]; dd = dd < 63 ? dd : 63;
  Hh[idx] += z_in[di * H_ + c] + z_out[dd * H_ + c];
}

__global__ void edge_dots_k(const float* __restrict__ ea, const float* __restrict__ We,
                            const float* __restrict__ be, const float* __restrict__ ev,
                            float* __restrict__ dots) {
  int e = blockIdx.x * 256 + threadIdx.x; if (e >= E_) return;
  float em[32];
#pragma unroll
  for (int d = 0; d < 32; ++d) em[d] = be[d];
#pragma unroll
  for (int f = 0; f < 16; ++f) {
    float a = ea[e * 16 + f];
#pragma unroll
    for (int d = 0; d < 32; ++d) em[d] += a * We[f * 32 + d];
  }
#pragma unroll
  for (int j = 0; j < MAXD_; ++j) {
    float s = 0.f;
#pragma unroll
    for (int d = 0; d < 32; ++d) s += em[d] * ev[j * 32 + d];
    dots[(size_t)e * MAXD_ + j] = s;
  }
}

__global__ void build_bias_k(const int* __restrict__ np, const int* __restrict__ ep,
                             const float* __restrict__ bsp, const float* __restrict__ dots,
                             float* __restrict__ bias) {
  int p = blockIdx.x * 256 + threadIdx.x; if (p >= P_) return;
  int npl = 0;
#pragma unroll
  for (int j = 0; j < MAXD_; ++j) npl += (np[(size_t)p * MAXD_ + j] != -1);
  float sp = (npl > 0) ? bsp[(npl < MAXD_ ? npl : MAXD_) - 1] : 0.f;
  int cnt = 0; float s = 0.f;
#pragma unroll
  for (int j = 0; j < MAXD_; ++j) {
    int e = ep[(size_t)p * MAXD_ + j];
    if (e != -1) { s += dots[(size_t)e * MAXD_ + j]; cnt++; }
  }
  float c = (cnt > 0) ? s / (float)cnt : 0.f;
  bias[p] = sp + c;
}

__global__ void build_xf_k(const float* __restrict__ x, __bf16* __restrict__ xf) {
  int idx = blockIdx.x * 256 + threadIdx.x;          // M_*F_ threads
  int node = idx >> 7, f = idx & 127;
  int g = node >> 7, r = node & 127;
  float v = (r == 0) ? -1.0f : x[(size_t)(g * 127 + r - 1) * F_ + f];
  xf[idx] = f2bf(v);
}

// src W[K,N] row-major f32  ->  dst Wt[N,K] bf16
__global__ void w_to_bf16_t(const float* __restrict__ W, __bf16* __restrict__ Wt,
                            int Kd, int Nd) {
  int idx = blockIdx.x * 256 + threadIdx.x;
  if (idx >= Kd * Nd) return;
  int n = idx / Kd, k = idx % Kd;
  Wt[idx] = f2bf(W[(size_t)k * Nd + n]);
}

__global__ void out_proj_k(const float* __restrict__ Hh, const float* __restrict__ Wout,
                           const float* __restrict__ bout, float* __restrict__ out) {
  int g = blockIdx.x;
  int lane = threadIdx.x;
  const float* hr = Hh + (size_t)g * N1_ * H_;       // virtual-node row
  float acc = 0.f;
  for (int i = lane; i < H_; i += 32) acc += hr[i] * Wout[i];
#pragma unroll
  for (int o = 16; o >= 1; o >>= 1) acc += __shfl_xor(acc, o, 32);
  if (lane == 0) out[g] = acc + bout[0];
}

// ---------------------------------------------------------------------------
extern "C" void kernel_launch(void* const* d_in, const int* in_sizes, int n_in,
                              void* d_out, int out_size, void* d_ws, size_t ws_size,
                              hipStream_t stream)
{
  const float* x    = (const float*)d_in[0];
  const int*   ei   = (const int*)  d_in[1];
  const float* ea   = (const float*)d_in[2];
  const int*   np   = (const int*)  d_in[4];
  const int*   ep   = (const int*)  d_in[5];
  const float* Wn   = (const float*)d_in[6];
  const float* bnv  = (const float*)d_in[7];
  const float* We   = (const float*)d_in[8];
  const float* bev  = (const float*)d_in[9];
  const float* z_in = (const float*)d_in[10];
  const float* z_out= (const float*)d_in[11];
  const float* bsp  = (const float*)d_in[12];
  const float* ev   = (const float*)d_in[13];
  const float* ln1s = (const float*)d_in[14];
  const float* ln1b = (const float*)d_in[15];
  const float* Wq   = (const float*)d_in[16];
  const float* bq   = (const float*)d_in[17];
  const float* Wk   = (const float*)d_in[18];
  const float* bk   = (const float*)d_in[19];
  const float* Wv   = (const float*)d_in[20];
  const float* bv   = (const float*)d_in[21];
  const float* Wo   = (const float*)d_in[22];
  const float* bo   = (const float*)d_in[23];
  const float* ln2s = (const float*)d_in[24];
  const float* ln2b = (const float*)d_in[25];
  const float* W1   = (const float*)d_in[26];
  const float* b1   = (const float*)d_in[27];
  const float* W2   = (const float*)d_in[28];
  const float* b2   = (const float*)d_in[29];
  const float* Wout = (const float*)d_in[30];
  const float* bout = (const float*)d_in[31];

  // ---- workspace carve-out (~160 MB) ----
  char* ws = (char*)d_ws;
  size_t off = 0;
  auto take = [&](size_t bytes) -> char* {
    char* p = ws + off; off += (bytes + 255) & ~(size_t)255; return p;
  };
  float*  Hbuf = (float*) take((size_t)M_ * H_ * 4);         // residual stream (f32)
  __bf16* Ybuf = (__bf16*)take((size_t)M_ * H_ * 2);         // LN output (bf16)
  char*   R    =          take((size_t)4 * M_ * H_ * 2);     // 64 MB shared region
  __bf16* Qb  = (__bf16*)R;
  __bf16* Kbf = (__bf16*)(R + (size_t)1 * M_ * H_ * 2);
  __bf16* Vt  = (__bf16*)(R + (size_t)2 * M_ * H_ * 2);      // transposed [H][M]
  __bf16* Ob  = (__bf16*)(R + (size_t)3 * M_ * H_ * 2);
  __bf16* Ub  = (__bf16*)R;                                  // FFN mid reuses Q/K/V/O
  float*  Bias = (float*)take((size_t)P_ * 4);
  float*  Dots = (float*)take((size_t)E_ * MAXD_ * 4);
  int*    InD  = (int*)  take((size_t)M_ * 4);
  int*    OutD = (int*)  take((size_t)M_ * 4);
  __bf16* Xf   = (__bf16*)take((size_t)M_ * F_ * 2);
  __bf16* WnT  = (__bf16*)take((size_t)H_ * F_ * 2);
  __bf16* WqT  = (__bf16*)take((size_t)L_ * H_ * H_ * 2);
  __bf16* WkT  = (__bf16*)take((size_t)L_ * H_ * H_ * 2);
  __bf16* WvT  = (__bf16*)take((size_t)L_ * H_ * H_ * 2);
  __bf16* WoT  = (__bf16*)take((size_t)L_ * H_ * H_ * 2);
  __bf16* W1T  = (__bf16*)take((size_t)L_ * H_ * FFN_ * 2);
  __bf16* W2T  = (__bf16*)take((size_t)L_ * FFN_ * H_ * 2);

  // ---- weights -> transposed bf16 ----
  w_to_bf16_t<<<(F_ * H_ + 255) / 256, 256, 0, stream>>>(Wn, WnT, F_, H_);
  for (int l = 0; l < L_; ++l) {
    w_to_bf16_t<<<(H_ * H_ + 255) / 256, 256, 0, stream>>>(Wq + (size_t)l * H_ * H_,   WqT + (size_t)l * H_ * H_,   H_, H_);
    w_to_bf16_t<<<(H_ * H_ + 255) / 256, 256, 0, stream>>>(Wk + (size_t)l * H_ * H_,   WkT + (size_t)l * H_ * H_,   H_, H_);
    w_to_bf16_t<<<(H_ * H_ + 255) / 256, 256, 0, stream>>>(Wv + (size_t)l * H_ * H_,   WvT + (size_t)l * H_ * H_,   H_, H_);
    w_to_bf16_t<<<(H_ * H_ + 255) / 256, 256, 0, stream>>>(Wo + (size_t)l * H_ * H_,   WoT + (size_t)l * H_ * H_,   H_, H_);
    w_to_bf16_t<<<(H_ * FFN_ + 255) / 256, 256, 0, stream>>>(W1 + (size_t)l * H_ * FFN_, W1T + (size_t)l * FFN_ * H_, H_, FFN_);
    w_to_bf16_t<<<(FFN_ * H_ + 255) / 256, 256, 0, stream>>>(W2 + (size_t)l * FFN_ * H_, W2T + (size_t)l * H_ * FFN_, FFN_, H_);
  }

  // ---- node features (vnode = -1 rows) + embedding GEMM ----
  build_xf_k<<<(M_ * F_) / 256, 256, 0, stream>>>(x, Xf);
  gemm_wmma<0><<<dim3(M_ / 64, H_ / 64), 256, 0, stream>>>(Xf, WnT, bnv, Hbuf, nullptr, M_, H_, F_);

  // ---- degree embeddings ----
  zero_i32<<<M_ / 256, 256, 0, stream>>>(InD,  M_);
  zero_i32<<<M_ / 256, 256, 0, stream>>>(OutD, M_);
  degrees_k<<<E_ / 256, 256, 0, stream>>>(ei, InD, OutD);
  add_deg_emb<<<(M_ * H_) / 256, 256, 0, stream>>>(Hbuf, InD, OutD, z_in, z_out);

  // ---- structural bias ----
  edge_dots_k<<<E_ / 256, 256, 0, stream>>>(ea, We, bev, ev, Dots);
  build_bias_k<<<P_ / 256, 256, 0, stream>>>(np, ep, bsp, Dots, Bias);

  // ---- transformer layers ----
  for (int l = 0; l < L_; ++l) {
    const size_t wo = (size_t)l * H_ * H_;
    layernorm_bf16<<<M_ / 8, 256, 0, stream>>>(Hbuf, ln1s + l * H_, ln1b + l * H_, Ybuf);
    gemm_wmma<1><<<dim3(M_ / 64, H_ / 64), 256, 0, stream>>>(Ybuf, WqT + wo, bq + l * H_, nullptr, Qb,  M_, H_, H_);
    gemm_wmma<1><<<dim3(M_ / 64, H_ / 64), 256, 0, stream>>>(Ybuf, WkT + wo, bk + l * H_, nullptr, Kbf, M_, H_, H_);
    gemm_wmma<2><<<dim3(M_ / 64, H_ / 64), 256, 0, stream>>>(Ybuf, WvT + wo, bv + l * H_, nullptr, Vt,  M_, H_, H_);
    attn_kernel<<<dim3(G_, NH_), 128, 0, stream>>>(Qb, Kbf, Vt, Bias, Ob);
    gemm_wmma<3><<<dim3(M_ / 64, H_ / 64), 256, 0, stream>>>(Ob, WoT + wo, bo + l * H_, Hbuf, nullptr, M_, H_, H_);
    layernorm_bf16<<<M_ / 8, 256, 0, stream>>>(Hbuf, ln2s + l * H_, ln2b + l * H_, Ybuf);
    gemm_wmma<4><<<dim3(M_ / 64, FFN_ / 64), 256, 0, stream>>>(Ybuf, W1T + (size_t)l * FFN_ * H_, b1 + l * FFN_, nullptr, Ub, M_, FFN_, H_);
    gemm_wmma<3><<<dim3(M_ / 64, H_ / 64), 256, 0, stream>>>(Ub, W2T + (size_t)l * H_ * FFN_, b2 + l * H_, Hbuf, nullptr, M_, H_, FFN_);
  }

  // ---- readout ----
  out_proj_k<<<G_, 32, 0, stream>>>(Hbuf, Wout, bout, (float*)d_out);
}
